// Encoder_54571854463117
// MI455X (gfx1250) — matrix operand order repaired and checked by reading, not compiled, not run on hardware
//
#include <hip/hip_runtime.h>
#include <hip/hip_bf16.h>
#include <math.h>

// ---------------- problem constants ----------------
#define B_      16
#define M_      2048
#define NC_     512
#define KN_     32
#define DIM_    384
#define HEADS_  6
#define HD_     64
#define FF_     1536
#define QKVD_   1152
#define FEAT_   51
#define FPAD_   64
#define C1_     256
#define ROWS_   (B_*NC_)   // 8192 tokens

typedef __attribute__((ext_vector_type(16))) _Float16 v16h;
typedef __attribute__((ext_vector_type(8)))  float    v8f;

// ---------------- WMMA fragment helpers (wave32, 16x16x32 f16) ----------------
// A: 16x32 (MxK) row-major tile in LDS. Lane L<16 holds M=L, K {0..15};
// lanes 16..31 hold K {16..31}. -> two contiguous 16B chunks.
static __device__ __forceinline__ v16h load_a_frag(const _Float16* p, int ld) {
  int lane = threadIdx.x & 31;
  int m  = lane & 15;
  int kb = (lane >> 4) << 4;
  v16h a;
#pragma unroll
  for (int i = 0; i < 8; ++i) {
    a[i]     = p[m * ld + kb + i];
    a[i + 8] = p[m * ld + kb + i + 8];
  }
  return a;
}

// B fragments from PRE-SWIZZLED weights: tile (kt,nt) of a KxN matrix is stored
// as 32 lanes x 16 halves contiguous -> one v16h (2x global_load_b128) per lane.
static __device__ __forceinline__ v16h load_bsw(const _Float16* w, int ntiles,
                                                int kt, int nt) {
  int lane = threadIdx.x & 31;
  return ((const v16h*)w)[((size_t)kt * ntiles + nt) * 32 + lane];
}

// B fragment where logical B(k,n) = p[n*ld + k]  (k^T for attention: contiguous)
static __device__ __forceinline__ v16h load_bT_frag(const _Float16* p, int ld) {
  int lane = threadIdx.x & 31;
  int n  = lane & 15;
  int kb = (lane >> 4) << 4;
  v16h b;
#pragma unroll
  for (int i = 0; i < 16; ++i) b[i] = p[n * ld + kb + i];
  return b;
}

static __device__ __forceinline__ v8f wmma16(v16h a, v16h b, v8f c) {
  return __builtin_amdgcn_wmma_f32_16x16x32_f16(false, a, false, b, (short)0, c,
                                                false, false);
}
// C layout: VGPR r of lane L -> m = r + 8*(L>>4), n = L&15  (ISA 7.12.2)

// ---- async global->LDS staging (GLOBAL_LOAD_ASYNC_TO_LDS_B128, ASYNCcnt) ----
// Copies rows x cols halves (cols % 8 == 0) from global (row stride srcStride
// halves) into a dense row-major LDS tile. Caller must __syncthreads() after.
static __device__ __forceinline__ void stage_async(_Float16* lds, const _Float16* src,
                                                   int rows, int cols, size_t srcStride) {
  uint32_t base = (uint32_t)(size_t)lds;   // low 32 bits = LDS byte offset
  int cpr = cols >> 3;                     // 16B chunks per row
  int chunks = rows * cpr;
  for (int c = threadIdx.x; c < chunks; c += blockDim.x) {
    int r  = c / cpr;
    int k8 = (c - r * cpr) << 3;
    uint32_t lo = base + (uint32_t)(r * cols + k8) * 2u;
    unsigned long long ga = (unsigned long long)(size_t)(src + (size_t)r * srcStride + k8);
    asm volatile("global_load_async_to_lds_b128 %0, %1, off"
                 :: "v"(lo), "v"(ga) : "memory");
  }
  asm volatile("s_wait_asynccnt 0" ::: "memory");
}

// ---------------- weight convert f32 -> f16, WMMA-fragment swizzled ----------------
// dst layout: tiles (K/32)x(N/16); within a tile: lane-major, 16 halves/lane.
__global__ void cvt_swz_kernel(const float* __restrict__ src, _Float16* __restrict__ dst,
                               int srcRows, int K, int N) {
  int ntiles = N >> 4;
  int total = (K >> 5) * ntiles * 512;
  for (int i = blockIdx.x * blockDim.x + threadIdx.x; i < total;
       i += gridDim.x * blockDim.x) {
    int e    = i & 15;
    int lane = (i >> 4) & 31;
    int tile = i >> 9;
    int nt = tile % ntiles;
    int kt = tile / ntiles;
    int row = (kt << 5) + ((lane >> 4) << 4) + e;
    int col = (nt << 4) + (lane & 15);
    dst[i] = (_Float16)(row < srcRows ? src[(size_t)row * N + col] : 0.0f);
  }
}

// ---------------- farthest point sampling (1 block / batch) ----------------
__global__ void fps_kernel(const float* __restrict__ pc, float* __restrict__ centers) {
  __shared__ float px[M_], py[M_], pz[M_], dist[M_];
  __shared__ float rv[256];
  __shared__ int   ri[256];
  int b = blockIdx.x, tid = threadIdx.x;
  const float* p = pc + (size_t)b * M_ * 3;
  for (int i = tid; i < M_; i += 256) {
    px[i] = p[3 * i]; py[i] = p[3 * i + 1]; pz[i] = p[3 * i + 2];
  }
  __syncthreads();
  float cx = px[0], cy = py[0], cz = pz[0];
  if (tid == 0) {
    float* c = centers + (size_t)b * NC_ * 3;
    c[0] = cx; c[1] = cy; c[2] = cz;
  }
  for (int i = tid; i < M_; i += 256) {
    float dx = px[i] - cx, dy = py[i] - cy, dz = pz[i] - cz;
    dist[i] = dx * dx + dy * dy + dz * dz;
  }
  __syncthreads();
  for (int s = 1; s < NC_; ++s) {
    float best = -1.0f; int bi = 0;
    for (int i = tid; i < M_; i += 256) {
      float d = dist[i];
      if (d > best) { best = d; bi = i; }
    }
    rv[tid] = best; ri[tid] = bi;
    __syncthreads();
    for (int off = 128; off > 0; off >>= 1) {
      if (tid < off) {
        float ov = rv[tid + off]; int oi = ri[tid + off];
        if (ov > rv[tid] || (ov == rv[tid] && oi < ri[tid])) { rv[tid] = ov; ri[tid] = oi; }
      }
      __syncthreads();
    }
    int im = ri[0];
    cx = px[im]; cy = py[im]; cz = pz[im];
    if (tid == 0) {
      float* c = centers + ((size_t)b * NC_ + s) * 3;
      c[0] = cx; c[1] = cy; c[2] = cz;
    }
    __syncthreads();
    for (int i = tid; i < M_; i += 256) {
      float dx = px[i] - cx, dy = py[i] - cy, dz = pz[i] - cz;
      float d = dx * dx + dy * dy + dz * dz;
      if (d < dist[i]) dist[i] = d;
    }
    __syncthreads();
  }
}

// ---------------- KNN: 1 block / center, 32x argmin-extract ----------------
__global__ void knn_kernel(const float* __restrict__ pc, const float* __restrict__ centers,
                           int* __restrict__ nn) {
  __shared__ float d2[M_];
  __shared__ float rv[256];
  __shared__ int   ri[256];
  int row = blockIdx.x, tid = threadIdx.x;
  int b = row >> 9;
  const float* p = pc + (size_t)b * M_ * 3;
  float cx = centers[(size_t)row * 3], cy = centers[(size_t)row * 3 + 1],
        cz = centers[(size_t)row * 3 + 2];
  for (int i = tid; i < M_; i += 256) {
    float dx = p[3 * i] - cx, dy = p[3 * i + 1] - cy, dz = p[3 * i + 2] - cz;
    d2[i] = dx * dx + dy * dy + dz * dz;
  }
  __syncthreads();
  for (int s = 0; s < KN_; ++s) {
    float best = 3.4e38f; int bi = 0;
    for (int i = tid; i < M_; i += 256) {
      float d = d2[i];
      if (d < best) { best = d; bi = i; }
    }
    rv[tid] = best; ri[tid] = bi;
    __syncthreads();
    for (int off = 128; off > 0; off >>= 1) {
      if (tid < off) {
        float ov = rv[tid + off]; int oi = ri[tid + off];
        if (ov < rv[tid] || (ov == rv[tid] && oi < ri[tid])) { rv[tid] = ov; ri[tid] = oi; }
      }
      __syncthreads();
    }
    if (tid == 0) { nn[(size_t)row * KN_ + s] = ri[0]; d2[ri[0]] = 3.4e38f; }
    __syncthreads();
  }
}

// ---------------- edge MLP: feat(51)->256 relu ->256, max over 32 neighbors ----------------
__global__ void edge_kernel(const float* __restrict__ pc, const float* __restrict__ centers,
                            const int* __restrict__ nn,
                            const _Float16* __restrict__ w1, const float* __restrict__ b1,
                            const _Float16* __restrict__ w2, const float* __restrict__ b2,
                            _Float16* __restrict__ hmax) {
  __shared__ _Float16 feat[KN_][FPAD_];   // 32 edges x 64 (51 real + zero pad)
  __shared__ _Float16 h1s[KN_][C1_];      // relu(feat@w1+b1)
  int row = blockIdx.x, tid = threadIdx.x;
  int b = row >> 9;
  float cx = centers[(size_t)row * 3], cy = centers[(size_t)row * 3 + 1],
        cz = centers[(size_t)row * 3 + 2];
  const float PI = 3.14159265358979323846f;
  int e = tid >> 3, fl = tid & 7;
  int ni = nn[(size_t)row * KN_ + e];
  const float* q = pc + ((size_t)b * M_ + ni) * 3;
  float rx = q[0] - cx, ry = q[1] - cy, rz = q[2] - cz;
#pragma unroll
  for (int t = 0; t < 8; ++t) {
    int f = fl + 8 * t;
    float v = 0.0f;
    if (f < 3) v = (f == 0) ? rx : ((f == 1) ? ry : rz);
    else if (f < 27) {
      int j = f - 3; float r = (j < 8) ? rx : ((j < 16) ? ry : rz);
      v = sinf(r * (PI * (float)(1 << (j & 7))));
    } else if (f < 51) {
      int j = f - 27; float r = (j < 8) ? rx : ((j < 16) ? ry : rz);
      v = cosf(r * (PI * (float)(1 << (j & 7))));
    }
    feat[e][f] = (_Float16)v;
  }
  __syncthreads();

  int wave = tid >> 5, lane = tid & 31;
  int n0 = wave * 32;
  int nt0 = n0 >> 4;
  v8f acc[2][2] = {};
#pragma unroll
  for (int ks = 0; ks < 2; ++ks) {
    int k0 = ks * 32;
    v16h a0  = load_a_frag(&feat[0][k0],  FPAD_);
    v16h a1  = load_a_frag(&feat[16][k0], FPAD_);
    v16h bb0 = load_bsw(w1, C1_ / 16, ks, nt0);
    v16h bb1 = load_bsw(w1, C1_ / 16, ks, nt0 + 1);
    acc[0][0] = wmma16(a0, bb0, acc[0][0]);
    acc[0][1] = wmma16(a0, bb1, acc[0][1]);
    acc[1][0] = wmma16(a1, bb0, acc[1][0]);
    acc[1][1] = wmma16(a1, bb1, acc[1][1]);
  }
#pragma unroll
  for (int mt = 0; mt < 2; ++mt)
#pragma unroll
    for (int nt = 0; nt < 2; ++nt)
#pragma unroll
      for (int r = 0; r < 8; ++r) {
        int m = mt * 16 + r + ((lane >> 4) << 3);
        int n = n0 + nt * 16 + (lane & 15);
        float v = acc[mt][nt][r] + b1[n];
        h1s[m][n] = (_Float16)(v > 0.0f ? v : 0.0f);
      }
  __syncthreads();

  v8f acc2[2][2] = {};
#pragma unroll
  for (int ks = 0; ks < 8; ++ks) {
    int k0 = ks * 32;
    v16h a0  = load_a_frag(&h1s[0][k0],  C1_);
    v16h a1  = load_a_frag(&h1s[16][k0], C1_);
    v16h bb0 = load_bsw(w2, C1_ / 16, ks, nt0);
    v16h bb1 = load_bsw(w2, C1_ / 16, ks, nt0 + 1);
    acc2[0][0] = wmma16(a0, bb0, acc2[0][0]);
    acc2[0][1] = wmma16(a0, bb1, acc2[0][1]);
    acc2[1][0] = wmma16(a1, bb0, acc2[1][0]);
    acc2[1][1] = wmma16(a1, bb1, acc2[1][1]);
  }
  // max over 32 neighbor rows: in-register + cross-half shuffle (wave32)
#pragma unroll
  for (int nt = 0; nt < 2; ++nt) {
    float mv = -3.4e38f;
#pragma unroll
    for (int r = 0; r < 8; ++r)
      mv = fmaxf(mv, fmaxf(acc2[0][nt][r], acc2[1][nt][r]));
    mv = fmaxf(mv, __shfl_xor(mv, 16, 32));
    int n = n0 + nt * 16 + (lane & 15);
    if (lane < 16) hmax[(size_t)row * C1_ + n] = (_Float16)(mv + b2[n]);
  }
}

// ---------------- global MLP 256->256 relu ->384 + fused PE ----------------
__global__ void gmlp_kernel(const _Float16* __restrict__ hmax, const float* __restrict__ centers,
                            const _Float16* __restrict__ gw1f, const float* __restrict__ gb1,
                            const _Float16* __restrict__ gw2f, const float* __restrict__ gb2,
                            const float* __restrict__ embed_w, const float* __restrict__ embed_b,
                            float* __restrict__ x) {
  __shared__ _Float16 hA[32][C1_];
  __shared__ _Float16 tS[32][C1_];
  __shared__ float    cf[32][FEAT_];
  int row0 = blockIdx.x * 32, tid = threadIdx.x;
  const float PI = 3.14159265358979323846f;
  stage_async(&hA[0][0], hmax + (size_t)row0 * C1_, 32, C1_, C1_);
  for (int i = tid; i < 32 * FEAT_; i += 256) {
    int m = i / FEAT_, f = i % FEAT_;
    const float* c = centers + (size_t)(row0 + m) * 3;
    float v;
    if (f < 3) v = c[f];
    else if (f < 27) { int j = f - 3;  v = sinf(c[j >> 3] * (PI * (float)(1 << (j & 7)))); }
    else             { int j = f - 27; v = cosf(c[j >> 3] * (PI * (float)(1 << (j & 7)))); }
    cf[m][f] = v;
  }
  __syncthreads();

  int wave = tid >> 5, lane = tid & 31;
  int n0 = wave * 32;
  int nt0 = n0 >> 4;
  v8f acc[2][2] = {};
  for (int ks = 0; ks < 8; ++ks) {
    int k0 = ks * 32;
    v16h a0  = load_a_frag(&hA[0][k0],  C1_);
    v16h a1  = load_a_frag(&hA[16][k0], C1_);
    v16h bb0 = load_bsw(gw1f, C1_ / 16, ks, nt0);
    v16h bb1 = load_bsw(gw1f, C1_ / 16, ks, nt0 + 1);
    acc[0][0] = wmma16(a0, bb0, acc[0][0]);
    acc[0][1] = wmma16(a0, bb1, acc[0][1]);
    acc[1][0] = wmma16(a1, bb0, acc[1][0]);
    acc[1][1] = wmma16(a1, bb1, acc[1][1]);
  }
#pragma unroll
  for (int mt = 0; mt < 2; ++mt)
#pragma unroll
    for (int nt = 0; nt < 2; ++nt)
#pragma unroll
      for (int r = 0; r < 8; ++r) {
        int m = mt * 16 + r + ((lane >> 4) << 3);
        int n = n0 + nt * 16 + (lane & 15);
        float v = acc[mt][nt][r] + gb1[n];
        tS[m][n] = (_Float16)(v > 0.0f ? v : 0.0f);
      }
  __syncthreads();

  int c0 = wave * 48;                       // 384 cols over 8 waves
  v8f a2[2][3] = {};
  for (int ks = 0; ks < 8; ++ks) {
    int k0 = ks * 32;
    v16h a0 = load_a_frag(&tS[0][k0],  C1_);
    v16h a1 = load_a_frag(&tS[16][k0], C1_);
#pragma unroll
    for (int st = 0; st < 3; ++st) {
      v16h bb = load_bsw(gw2f, DIM_ / 16, ks, (c0 >> 4) + st);
      a2[0][st] = wmma16(a0, bb, a2[0][st]);
      a2[1][st] = wmma16(a1, bb, a2[1][st]);
    }
  }
#pragma unroll
  for (int mt = 0; mt < 2; ++mt)
#pragma unroll
    for (int st = 0; st < 3; ++st)
#pragma unroll
      for (int r = 0; r < 8; ++r) {
        int m = mt * 16 + r + ((lane >> 4) << 3);
        int n = c0 + st * 16 + (lane & 15);
        float v = a2[mt][st][r] + gb2[n] + embed_b[n];
        for (int f = 0; f < FEAT_; ++f) v += cf[m][f] * embed_w[f * DIM_ + n];
        x[(size_t)(row0 + m) * DIM_ + n] = v;
      }
}

// ---------------- layernorm (f16 out for GEMM A, or f32 final out) ----------------
__global__ void ln_kernel(const float* __restrict__ x, const float* __restrict__ g,
                          const float* __restrict__ bta,
                          _Float16* __restrict__ o16, float* __restrict__ o32) {
  __shared__ float red[128];
  int row = blockIdx.x, tid = threadIdx.x;
  const float* xr = x + (size_t)row * DIM_;
  float s = 0.0f;
  for (int i = tid; i < DIM_; i += 128) s += xr[i];
  red[tid] = s; __syncthreads();
  for (int off = 64; off > 0; off >>= 1) { if (tid < off) red[tid] += red[tid + off]; __syncthreads(); }
  float mean = red[0] * (1.0f / DIM_);
  __syncthreads();
  float s2 = 0.0f;
  for (int i = tid; i < DIM_; i += 128) { float d = xr[i] - mean; s2 += d * d; }
  red[tid] = s2; __syncthreads();
  for (int off = 64; off > 0; off >>= 1) { if (tid < off) red[tid] += red[tid + off]; __syncthreads(); }
  float rstd = rsqrtf(red[0] * (1.0f / DIM_) + 1e-6f);
  for (int i = tid; i < DIM_; i += 128) {
    float v = (xr[i] - mean) * rstd * g[i] + bta[i];
    if (o16) o16[(size_t)row * DIM_ + i] = (_Float16)v;
    else     o32[(size_t)row * DIM_ + i] = v;
  }
}

// ---------------- qkv GEMM; q/k stored [B,H,N,64], v stored WMMA-B-swizzled ----------------
__global__ void qkv_kernel(const _Float16* __restrict__ xn, const _Float16* __restrict__ w,
                           const float* __restrict__ bias,
                           _Float16* __restrict__ qf, _Float16* __restrict__ kf,
                           _Float16* __restrict__ vf) {
  __shared__ _Float16 A[32][DIM_];
  int row0 = blockIdx.x * 32, tid = threadIdx.x;
  stage_async(&A[0][0], xn + (size_t)row0 * DIM_, 32, DIM_, DIM_);
  __syncthreads();
  int wave = tid >> 5, lane = tid & 31;
  for (int grp = 0; grp < 3; ++grp) {
    int c0 = grp * DIM_ + wave * 48;
    v8f acc[2][3] = {};
    for (int ks = 0; ks < 12; ++ks) {
      int k0 = ks * 32;
      v16h a0 = load_a_frag(&A[0][k0],  DIM_);
      v16h a1 = load_a_frag(&A[16][k0], DIM_);
#pragma unroll
      for (int st = 0; st < 3; ++st) {
        v16h bb = load_bsw(w, QKVD_ / 16, ks, (c0 >> 4) + st);
        acc[0][st] = wmma16(a0, bb, acc[0][st]);
        acc[1][st] = wmma16(a1, bb, acc[1][st]);
      }
    }
#pragma unroll
    for (int mt = 0; mt < 2; ++mt)
#pragma unroll
      for (int st = 0; st < 3; ++st)
#pragma unroll
        for (int r = 0; r < 8; ++r) {
          int m = mt * 16 + r + ((lane >> 4) << 3);
          int j = c0 + st * 16 + (lane & 15);
          float v = acc[mt][st][r] + bias[j];
          int three = j / DIM_;
          int h = (j - three * DIM_) >> 6;
          int d = j & 63;
          int grow = row0 + m;
          int bidx = grow >> 9, tok = grow & 511;
          size_t bh = (size_t)bidx * HEADS_ + h;
          if (three < 2) {
            _Float16* dst = (three == 0) ? qf : kf;
            dst[(bh * NC_ + tok) * HD_ + d] = (_Float16)v;
          } else {
            // store V in WMMA B-fragment order for the P@V GEMM (K=512, N=64)
            int kt = tok >> 5;
            int ee = tok & 15;
            int ln = (((tok >> 4) & 1) << 4) + (d & 15);
            int nt = d >> 4;
            vf[bh * (NC_ * HD_) + (size_t)(((kt << 2) + nt) * 32 + ln) * 16 + ee] =
                (_Float16)v;
          }
        }
  }
}

// ---------------- attention: per (b,h,16-row q tile) ----------------
__global__ void attn_kernel(const _Float16* __restrict__ qf, const _Float16* __restrict__ kf,
                            const _Float16* __restrict__ vf, _Float16* __restrict__ of) {
  __shared__ _Float16 qs[16][HD_];
  __shared__ float    S[16][NC_];
  __shared__ _Float16 P[16][NC_];
  __shared__ float    rinv[16];
  int blk = blockIdx.x, tid = threadIdx.x;
  int qt = blk & 31, bh = blk >> 5;
  const _Float16* qrow = qf + ((size_t)bh * NC_ + qt * 16) * HD_;
  const _Float16* kb   = kf + (size_t)bh * NC_ * HD_;
  const _Float16* vb   = vf + (size_t)bh * NC_ * HD_;   // swizzled fragments
  stage_async(&qs[0][0], qrow, 16, HD_, HD_);
  __syncthreads();
  int wave = tid >> 5, lane = tid & 31;
#pragma unroll
  for (int st = 0; st < 8; ++st) {
    int n0 = wave * 128 + st * 16;
    v8f acc = {};
#pragma unroll
    for (int ks = 0; ks < 2; ++ks) {
      v16h a  = load_a_frag(&qs[0][ks * 32], HD_);
      v16h bb = load_bT_frag(kb + (size_t)n0 * HD_ + ks * 32, HD_); // B(k,n)=k[n][k]
      acc = wmma16(a, bb, acc);
    }
#pragma unroll
    for (int r = 0; r < 8; ++r) {
      int m = r + ((lane >> 4) << 3);
      int n = n0 + (lane & 15);
      S[m][n] = acc[r] * 0.125f;             // hd^-0.5
    }
  }
  __syncthreads();
  if (tid < 16) {
    float mx = -3.4e38f;
    for (int j = 0; j < NC_; ++j) mx = fmaxf(mx, S[tid][j]);
    float sum = 0.0f;
    for (int j = 0; j < NC_; ++j) { float ev = expf(S[tid][j] - mx); S[tid][j] = ev; sum += ev; }
    rinv[tid] = 1.0f / sum;
  }
  __syncthreads();
  for (int i = tid; i < 16 * NC_; i += 128)
    P[i >> 9][i & 511] = (_Float16)(S[i >> 9][i & 511] * rinv[i >> 9]);
  __syncthreads();
  v8f acc = {};
#pragma unroll
  for (int ks = 0; ks < 16; ++ks) {
    v16h a  = load_a_frag(&P[0][ks * 32], NC_);
    v16h bb = ((const v16h*)vb)[(size_t)(((ks << 2) + wave) * 32 + lane)];
    acc = wmma16(a, bb, acc);
  }
  int bidx = bh / HEADS_, h = bh % HEADS_;
#pragma unroll
  for (int r = 0; r < 8; ++r) {
    int m = r + ((lane >> 4) << 3);
    int n = wave * 16 + (lane & 15);
    of[((size_t)bidx * NC_ + qt * 16 + m) * DIM_ + h * HD_ + n] = (_Float16)acc[r];
  }
}

// ---------------- C = A[8192xK] @ W[Kx384] + bias; x += gamma*C ----------------
__global__ void gemm_res_kernel(const _Float16* __restrict__ A, int K,
                                const _Float16* __restrict__ W, const float* __restrict__ bias,
                                const float* __restrict__ gamma, float* __restrict__ x) {
  __shared__ _Float16 As[32][DIM_];
  int row0 = blockIdx.x * 32, tid = threadIdx.x;
  int wave = tid >> 5, lane = tid & 31;
  int c0 = wave * 48;
  v8f acc[2][3] = {};
  int nch = K / DIM_;
  for (int ch = 0; ch < nch; ++ch) {
    stage_async(&As[0][0], A + (size_t)row0 * K + ch * DIM_, 32, DIM_, K);
    __syncthreads();
    for (int ks = 0; ks < 12; ++ks) {
      int k0 = ks * 32;
      int kt = ch * 12 + ks;
      v16h a0 = load_a_frag(&As[0][k0],  DIM_);
      v16h a1 = load_a_frag(&As[16][k0], DIM_);
#pragma unroll
      for (int st = 0; st < 3; ++st) {
        v16h bb = load_bsw(W, DIM_ / 16, kt, (c0 >> 4) + st);
        acc[0][st] = wmma16(a0, bb, acc[0][st]);
        acc[1][st] = wmma16(a1, bb, acc[1][st]);
      }
    }
    __syncthreads();
  }
#pragma unroll
  for (int mt = 0; mt < 2; ++mt)
#pragma unroll
    for (int st = 0; st < 3; ++st)
#pragma unroll
      for (int r = 0; r < 8; ++r) {
        int m = mt * 16 + r + ((lane >> 4) << 3);
        int n = c0 + st * 16 + (lane & 15);
        float v = acc[mt][st][r] + bias[n];
        size_t idx = (size_t)(row0 + m) * DIM_ + n;
        x[idx] += gamma[n] * v;
      }
}

// ---------------- fc1: xn[8192x384] @ W[384x1536] + b, exact GELU, f16 out ----------------
__global__ void fc1_kernel(const _Float16* __restrict__ xn, const _Float16* __restrict__ W,
                           const float* __restrict__ bias, _Float16* __restrict__ t) {
  __shared__ _Float16 As[32][DIM_];
  int row0 = blockIdx.x * 32, tid = threadIdx.x;
  stage_async(&As[0][0], xn + (size_t)row0 * DIM_, 32, DIM_, DIM_);
  __syncthreads();
  int wave = tid >> 5, lane = tid & 31;
  for (int grp = 0; grp < 4; ++grp) {
    int c0 = grp * DIM_ + wave * 48;
    v8f acc[2][3] = {};
    for (int ks = 0; ks < 12; ++ks) {
      int k0 = ks * 32;
      v16h a0 = load_a_frag(&As[0][k0],  DIM_);
      v16h a1 = load_a_frag(&As[16][k0], DIM_);
#pragma unroll
      for (int st = 0; st < 3; ++st) {
        v16h bb = load_bsw(W, FF_ / 16, ks, (c0 >> 4) + st);
        acc[0][st] = wmma16(a0, bb, acc[0][st]);
        acc[1][st] = wmma16(a1, bb, acc[1][st]);
      }
    }
#pragma unroll
    for (int mt = 0; mt < 2; ++mt)
#pragma unroll
      for (int st = 0; st < 3; ++st)
#pragma unroll
        for (int r = 0; r < 8; ++r) {
          int m = mt * 16 + r + ((lane >> 4) << 3);
          int n = c0 + st * 16 + (lane & 15);
          float v = acc[mt][st][r] + bias[n];
          v = 0.5f * v * (1.0f + erff(v * 0.70710678118654752f));  // exact GELU
          t[(size_t)(row0 + m) * FF_ + n] = (_Float16)v;
        }
  }
}

// ---------------- host driver ----------------
extern "C" void kernel_launch(void* const* d_in, const int* in_sizes, int n_in,
                              void* d_out, int out_size, void* d_ws, size_t ws_size,
                              hipStream_t stream) {
  (void)in_sizes; (void)n_in; (void)out_size; (void)ws_size;
  const float* pc      = (const float*)d_in[0];
  const float* embed_w = (const float*)d_in[1];
  const float* embed_b = (const float*)d_in[2];
  const float* lw1     = (const float*)d_in[3];
  const float* lb1     = (const float*)d_in[4];
  const float* lw2     = (const float*)d_in[5];
  const float* lb2     = (const float*)d_in[6];
  const float* gw1     = (const float*)d_in[7];
  const float* gb1     = (const float*)d_in[8];
  const float* gw2     = (const float*)d_in[9];
  const float* gb2     = (const float*)d_in[10];
  const float* ln1_g   = (const float*)d_in[11];
  const float* ln1_b   = (const float*)d_in[12];
  const float* qkv_w   = (const float*)d_in[13];
  const float* qkv_b   = (const float*)d_in[14];
  const float* proj_w  = (const float*)d_in[15];
  const float* proj_b  = (const float*)d_in[16];
  const float* gamma1  = (const float*)d_in[17];
  const float* ln2_g   = (const float*)d_in[18];
  const float* ln2_b   = (const float*)d_in[19];
  const float* fc1_w   = (const float*)d_in[20];
  const float* fc1_b   = (const float*)d_in[21];
  const float* fc2_w   = (const float*)d_in[22];
  const float* fc2_b   = (const float*)d_in[23];
  const float* gamma2  = (const float*)d_in[24];
  const float* lnf_g   = (const float*)d_in[25];
  const float* lnf_b   = (const float*)d_in[26];

  float* out     = (float*)d_out;
  float* centers = out + (size_t)ROWS_ * DIM_;   // tuple tail: [B,N,3]

  char* ws = (char*)d_ws;
  size_t off = 0;
  auto alloc = [&](size_t bytes) -> void* {
    void* p = (void*)(ws + off);
    off += (bytes + 255) & ~(size_t)255;
    return p;
  };
  _Float16* lw1p   = (_Float16*)alloc((size_t)FPAD_ * C1_ * 2);
  _Float16* lw2f   = (_Float16*)alloc((size_t)C1_ * C1_ * 2);
  _Float16* gw1f   = (_Float16*)alloc((size_t)C1_ * C1_ * 2);
  _Float16* gw2f   = (_Float16*)alloc((size_t)C1_ * DIM_ * 2);
  _Float16* qkvwf  = (_Float16*)alloc((size_t)6 * DIM_ * QKVD_ * 2);
  _Float16* projwf = (_Float16*)alloc((size_t)6 * DIM_ * DIM_ * 2);
  _Float16* fc1wf  = (_Float16*)alloc((size_t)6 * DIM_ * FF_ * 2);
  _Float16* fc2wf  = (_Float16*)alloc((size_t)6 * FF_ * DIM_ * 2);
  int*      nn     = (int*)alloc((size_t)ROWS_ * KN_ * 4);
  _Float16* hmax   = (_Float16*)alloc((size_t)ROWS_ * C1_ * 2);
  float*    x      = (float*)alloc((size_t)ROWS_ * DIM_ * 4);
  _Float16* xn     = (_Float16*)alloc((size_t)ROWS_ * DIM_ * 2);
  _Float16* qf     = (_Float16*)alloc((size_t)ROWS_ * DIM_ * 2);
  _Float16* kf     = (_Float16*)alloc((size_t)ROWS_ * DIM_ * 2);
  _Float16* vf     = (_Float16*)alloc((size_t)ROWS_ * DIM_ * 2);
  _Float16* of     = (_Float16*)alloc((size_t)ROWS_ * DIM_ * 2);
  _Float16* t      = (_Float16*)alloc((size_t)ROWS_ * FF_ * 2);

  auto cvts = [&](const float* s, _Float16* dd, int sr, int K, int N) {
    int n = (K >> 5) * (N >> 4) * 512;
    int blocks = (n + 255) / 256; if (blocks > 4096) blocks = 4096;
    cvt_swz_kernel<<<blocks, 256, 0, stream>>>(s, dd, sr, K, N);
  };

  fps_kernel<<<B_, 256, 0, stream>>>(pc, centers);
  knn_kernel<<<ROWS_, 256, 0, stream>>>(pc, centers, nn);

  cvts(lw1, lw1p, FEAT_, FPAD_, C1_);        // pad 51 -> 64 rows with zeros
  cvts(lw2, lw2f, C1_, C1_, C1_);
  cvts(gw1, gw1f, C1_, C1_, C1_);
  cvts(gw2, gw2f, C1_, C1_, DIM_);
  for (int l = 0; l < 6; ++l) {
    cvts(qkv_w  + (size_t)l * DIM_ * QKVD_, qkvwf  + (size_t)l * DIM_ * QKVD_, DIM_, DIM_, QKVD_);
    cvts(proj_w + (size_t)l * DIM_ * DIM_,  projwf + (size_t)l * DIM_ * DIM_,  DIM_, DIM_, DIM_);
    cvts(fc1_w  + (size_t)l * DIM_ * FF_,   fc1wf  + (size_t)l * DIM_ * FF_,   DIM_, DIM_, FF_);
    cvts(fc2_w  + (size_t)l * FF_ * DIM_,   fc2wf  + (size_t)l * FF_ * DIM_,   FF_,  FF_,  DIM_);
  }

  edge_kernel<<<ROWS_, 256, 0, stream>>>(pc, centers, nn, lw1p, lb1, lw2f, lb2, hmax);
  gmlp_kernel<<<ROWS_ / 32, 256, 0, stream>>>(hmax, centers, gw1f, gb1, gw2f, gb2,
                                              embed_w, embed_b, x);

  for (int l = 0; l < 6; ++l) {
    ln_kernel<<<ROWS_, 128, 0, stream>>>(x, ln1_g + l * DIM_, ln1_b + l * DIM_, xn, nullptr);
    qkv_kernel<<<ROWS_ / 32, 256, 0, stream>>>(xn, qkvwf + (size_t)l * DIM_ * QKVD_,
                                               qkv_b + l * QKVD_, qf, kf, vf);
    attn_kernel<<<B_ * HEADS_ * (NC_ / 16), 128, 0, stream>>>(qf, kf, vf, of);
    gemm_res_kernel<<<ROWS_ / 32, 256, 0, stream>>>(of, DIM_,
                                                    projwf + (size_t)l * DIM_ * DIM_,
                                                    proj_b + l * DIM_, gamma1 + l * DIM_, x);
    ln_kernel<<<ROWS_, 128, 0, stream>>>(x, ln2_g + l * DIM_, ln2_b + l * DIM_, xn, nullptr);
    fc1_kernel<<<ROWS_ / 32, 256, 0, stream>>>(xn, fc1wf + (size_t)l * DIM_ * FF_,
                                               fc1_b + l * FF_, t);
    gemm_res_kernel<<<ROWS_ / 32, 256, 0, stream>>>(t, FF_,
                                                    fc2wf + (size_t)l * FF_ * DIM_,
                                                    fc2_b + l * DIM_, gamma2 + l * DIM_, x);
  }
  ln_kernel<<<ROWS_, 128, 0, stream>>>(x, lnf_g, lnf_b, nullptr, out);
}
